// QkvRelativeLocalAttention_35270271434815
// MI455X (gfx1250) — matrix operand order repaired and checked by reading, not compiled, not run on hardware
//
#include <hip/hip_runtime.h>

typedef __attribute__((ext_vector_type(16))) _Float16 v16h;
typedef __attribute__((ext_vector_type(8)))  _Float16 v8h;
typedef __attribute__((ext_vector_type(8)))  float    v8f;

#define BATCH 2
#define SEQ   4096
#define HEADS 8
#define DHEAD 64
#define LBLK  128
#define NBLK  32
#define WWIN  255
#define RV    32

// LDS strides (in halfs / floats), padded to avoid bank conflicts
#define KT_STRIDE 72    // K tile rows [384][72] f16   (row = key, col = d)
#define VT_STRIDE 392   // V^T rows   [64][392] f16    (row = d,   col = key)
#define P_STRIDE  40    // P buffer   [16][40]  f16 per wave

__device__ __forceinline__ v16h ld16h(const _Float16* p) {
  // two 16B LDS loads (16B-aligned strides), combined into a v16h operand
  v8h lo = *(const v8h*)p;
  v8h h2 = *(const v8h*)(p + 8);
  v16h r;
#pragma unroll
  for (int e = 0; e < 8; ++e) { r[e] = lo[e]; r[8 + e] = h2[e]; }
  return r;
}

__device__ __forceinline__ float half_red_max(float v, int lane) {
#pragma unroll
  for (int m = 1; m < 16; m <<= 1) {
    int o = __builtin_amdgcn_ds_bpermute((((lane ^ m) & 31) << 2), __float_as_int(v));
    v = fmaxf(v, __int_as_float(o));
  }
  return v;
}

__device__ __forceinline__ float half_red_sum(float v, int lane) {
#pragma unroll
  for (int m = 1; m < 16; m <<= 1) {
    int o = __builtin_amdgcn_ds_bpermute((((lane ^ m) & 31) << 2), __float_as_int(v));
    v += __int_as_float(o);
  }
  return v;
}

__global__ void __launch_bounds__(256)
local_attn_kernel(const float* __restrict__ Q, const float* __restrict__ K,
                  const float* __restrict__ V, const int* __restrict__ AM,
                  const int* __restrict__ IDS, const float* __restrict__ EMB,
                  const float* __restrict__ BIAS, float* __restrict__ OUT) {
  extern __shared__ char smem[];
  _Float16* Kt = (_Float16*)smem;                          // [384][KT_STRIDE]
  _Float16* Vt = Kt + 384 * KT_STRIDE;                     // [64][VT_STRIDE]
  float*    Rl = (float*)(Vt + 64 * VT_STRIDE);            // [8 waves][16][32]
  _Float16* Pb = (_Float16*)(Rl + 8 * 16 * 32);            // [8 waves][16][P_STRIDE]

  const int h = blockIdx.x, n = blockIdx.y, b = blockIdx.z;
  const int tid  = threadIdx.x;
  const int wave = tid >> 5, lane = tid & 31, l16 = lane & 15, hi = lane >> 4;

  // ---- Phase A: stage K tile [384,64] and V^T tile [64,384] in LDS as f16 ----
  {
    const int base_key = n * LBLK - LBLK;  // (n-1)*L : [prev | cur | next]
#pragma unroll
    for (int it = 0; it < 24; ++it) {
      const int idx = it * 256 + tid;      // 0..6143, 4 elems each
      const int k3 = idx >> 4;
      const int d0 = (idx & 15) << 2;
      const int kp = base_key + k3;
      float4 kv = make_float4(0.f, 0.f, 0.f, 0.f);
      float4 vv = make_float4(0.f, 0.f, 0.f, 0.f);
      if (kp >= 0 && kp < SEQ) {
        const int g = ((b * SEQ + kp) * HEADS + h) * DHEAD + d0;
        kv = *(const float4*)(K + g);
        vv = *(const float4*)(V + g);
      }
      _Float16* kd = Kt + k3 * KT_STRIDE + d0;
      kd[0] = (_Float16)kv.x; kd[1] = (_Float16)kv.y;
      kd[2] = (_Float16)kv.z; kd[3] = (_Float16)kv.w;
      Vt[(d0 + 0) * VT_STRIDE + k3] = (_Float16)vv.x;
      Vt[(d0 + 1) * VT_STRIDE + k3] = (_Float16)vv.y;
      Vt[(d0 + 2) * VT_STRIDE + k3] = (_Float16)vv.z;
      Vt[(d0 + 3) * VT_STRIDE + k3] = (_Float16)vv.w;
    }
  }

  // ---- Phase B: Q rows -> A-operand registers (16x32 A, contiguous k = 16*hi + e order) ----
  v16h qa[2];
  {
    const int sq = n * LBLK + wave * 16 + l16;   // A layout: lane%16 holds row M
    const float* qp = Q + ((b * SEQ + sq) * HEADS + h) * DHEAD;
#pragma unroll
    for (int c = 0; c < 2; ++c) {
      const int b1 = 32 * c + hi * 16;           // K = b1+e       (e=0..7)
      const int b2 = 32 * c + hi * 16 + 8;       // K = b2+(e-8)   (e=8..15)
#pragma unroll
      for (int e = 0; e < 8; ++e) {
        qa[c][e]     = (_Float16)qp[b1 + e];
        qa[c][8 + e] = (_Float16)qp[b2 + e];
      }
    }
  }

  // all_rel[16,32] = Q[16,64] x emb^T[64,32] + bias  (per wave, into LDS)
  float* RlW = Rl + wave * 16 * 32;
  {
#pragma unroll
    for (int rt = 0; rt < 2; ++rt) {
      v8f racc = {};
#pragma unroll
      for (int c = 0; c < 2; ++c) {
        const float* ep = EMB + ((rt * 16 + l16) * HEADS + h) * DHEAD
                        + 32 * c + hi * 16;      // B layout: K = 32c+16*hi+e
        v16h bv;
#pragma unroll
        for (int e = 0; e < 16; ++e) bv[e] = (_Float16)ep[e];
        racc = __builtin_amdgcn_wmma_f32_16x16x32_f16(false, qa[c], false, bv,
                                                      (short)0, racc, false, false);
      }
      const float bias_v = BIAS[(rt * 16 + l16) * HEADS + h];
#pragma unroll
      for (int j = 0; j < 8; ++j)
        RlW[(hi * 8 + j) * 32 + rt * 16 + l16] = racc[j] + bias_v;
    }
  }
  __syncthreads();

  // ---- Hoisted per-row constants for the gather/score path (32-bit math) ----
  // row j handles query row qrow_j = wave*16 + hi*8 + j (C/D layout)
  int   row_base[8];       // (b*SEQ + q_abs)*WWIN - qrow - 1 ; + key3 = gather idx
  int   qrow_c[8];         // qrow + 1 (for window test)
  const float* relrow[8];  // &RlW[(hi*8+j)*32]
#pragma unroll
  for (int j = 0; j < 8; ++j) {
    const int qrow  = wave * 16 + hi * 8 + j;
    const int q_abs = n * LBLK + qrow;
    row_base[j] = (b * SEQ + q_abs) * WWIN - qrow - 1;
    qrow_c[j]   = qrow + 1;
    relrow[j]   = RlW + (hi * 8 + j) * 32;
  }

  // ---- Phase C: flash-attention loop over 12 supertiles of 32 keys ----
  v8f oacc[4];
  {
    v8f z = {};
#pragma unroll
    for (int t = 0; t < 4; ++t) oacc[t] = z;
  }
  float m_run[8], l_run[8];
#pragma unroll
  for (int j = 0; j < 8; ++j) { m_run[j] = -1e30f; l_run[j] = 0.f; }

  _Float16* Pw = Pb + wave * 16 * P_STRIDE;
  const int key_lo = LBLK - n * LBLK;                 // key3 >= key_lo  => key_abs >= 0
  const int key_hi = SEQ + LBLK - n * LBLK;           // key3 <  key_hi  => key_abs < SEQ

  for (int st = 0; st < 12; ++st) {
    // S tiles: content = Q x K^T for 2 x 16-key tiles
    v8f sacc[2];
#pragma unroll
    for (int t2 = 0; t2 < 2; ++t2) {
      v8f a = {};
      const int kb = st * 2 + t2;
#pragma unroll
      for (int c = 0; c < 2; ++c) {
        v16h bv = ld16h(Kt + (kb * 16 + l16) * KT_STRIDE + 32 * c + hi * 16);
        a = __builtin_amdgcn_wmma_f32_16x16x32_f16(false, qa[c], false, bv,
                                                   (short)0, a, false, false);
      }
      sacc[t2] = a;
    }

    // scores = (content + rel)/sqrt(d) - 10000*(1-mask)
    float sc[2][8];
#pragma unroll
    for (int t2 = 0; t2 < 2; ++t2) {
      const int key3  = (st * 2 + t2) * 16 + l16;     // C layout: col = lane%16
      const bool key_ok = (key3 >= key_lo) && (key3 < key_hi);
#pragma unroll
      for (int j = 0; j < 8; ++j) {
        const unsigned wdx = (unsigned)(key3 - qrow_c[j]);  // skew: w = k3 - q - 1
        const bool win_ok = wdx < (unsigned)WWIN;
        int mv = 0, idv = 0;
        if (win_ok) {
          const int mi = row_base[j] + key3;          // 32-bit, hoisted base
          mv  = AM[mi];
          idv = IDS[mi];
        }
        const bool ok = win_ok && key_ok && (mv != 0);
        const float rel = relrow[j][idv];
        float s = (sacc[t2][j] + rel) * 0.125f;       // 1/sqrt(64)
        if (!ok) s -= 10000.0f;
        sc[t2][j] = s;
      }
    }

    // online softmax update (row stats reduced over 16-lane halves)
    float p0[8], p1[8];
#pragma unroll
    for (int j = 0; j < 8; ++j) {
      float mt = fmaxf(sc[0][j], sc[1][j]);
      mt = half_red_max(mt, lane);
      const float mn   = fmaxf(m_run[j], mt);
      const float corr = __expf(m_run[j] - mn);
      m_run[j] = mn;
      p0[j] = __expf(sc[0][j] - mn);
      p1[j] = __expf(sc[1][j] - mn);
      l_run[j] = l_run[j] * corr + p0[j] + p1[j];
#pragma unroll
      for (int t = 0; t < 4; ++t) oacc[t][j] *= corr;
    }

    // P (C layout) -> LDS f16 -> reload in A-operand layout
#pragma unroll
    for (int j = 0; j < 8; ++j) {
      Pw[(hi * 8 + j) * P_STRIDE + l16]      = (_Float16)p0[j];
      Pw[(hi * 8 + j) * P_STRIDE + 16 + l16] = (_Float16)p1[j];
    }
    v16h pa;
    {
      v8h lo = *(const v8h*)(Pw + l16 * P_STRIDE + hi * 16);
      v8h h2 = *(const v8h*)(Pw + l16 * P_STRIDE + hi * 16 + 8);
#pragma unroll
      for (int e = 0; e < 8; ++e) { pa[e] = lo[e]; pa[8 + e] = h2[e]; }
    }

    // O += P[16,32] x V[32,64]  (B operand from V^T rows: contiguous key pairs)
#pragma unroll
    for (int t = 0; t < 4; ++t) {
      v16h bv = ld16h(Vt + (t * 16 + l16) * VT_STRIDE + st * 32 + hi * 16);
      oacc[t] = __builtin_amdgcn_wmma_f32_16x16x32_f16(false, pa, false, bv,
                                                       (short)0, oacc[t], false, false);
    }
  }

  // ---- Finalize: divide by row sums, store f32 output ----
#pragma unroll
  for (int j = 0; j < 8; ++j) {
    const float lr  = half_red_sum(l_run[j], lane);
    const float inv = 1.0f / lr;
    const int sq = n * LBLK + wave * 16 + hi * 8 + j;
    float* op = OUT + ((b * SEQ + sq) * HEADS + h) * DHEAD;
#pragma unroll
    for (int t = 0; t < 4; ++t)
      op[t * 16 + l16] = oacc[t][j] * inv;
  }
}

extern "C" void kernel_launch(void* const* d_in, const int* in_sizes, int n_in,
                              void* d_out, int out_size, void* d_ws, size_t ws_size,
                              hipStream_t stream) {
  const float* Q    = (const float*)d_in[0];
  const float* K    = (const float*)d_in[1];
  const float* V    = (const float*)d_in[2];
  const int*   AM   = (const int*)d_in[3];
  const int*   IDS  = (const int*)d_in[4];
  const float* EMB  = (const float*)d_in[5];
  const float* BIAS = (const float*)d_in[6];
  float* OUT = (float*)d_out;

  const size_t shmem = (size_t)(384 * KT_STRIDE + 64 * VT_STRIDE) * sizeof(_Float16)
                     + (size_t)(8 * 16 * 32) * sizeof(float)
                     + (size_t)(8 * 16 * P_STRIDE) * sizeof(_Float16);

  (void)hipFuncSetAttribute((const void*)local_attn_kernel,
                            hipFuncAttributeMaxDynamicSharedMemorySize, (int)shmem);

  dim3 grid(HEADS, NBLK, BATCH);
  local_attn_kernel<<<grid, 256, shmem, stream>>>(Q, K, V, AM, IDS, EMB, BIAS, OUT);
}
